// GraphEncoder_81827716924178
// MI455X (gfx1250) — matrix operand order repaired and checked by reading, not compiled, not run on hardware
//
#include <hip/hip_runtime.h>
#include <hip/hip_bf16.h>

typedef __attribute__((ext_vector_type(16))) __bf16 v16bf;
typedef __attribute__((ext_vector_type(8)))  float  v8f;
typedef unsigned int u32;

#define NHEADS 4

__device__ __forceinline__ unsigned short bf16r(float f){
    u32 u = __float_as_uint(f);
    u += 0x7fffu + ((u >> 16) & 1u);
    return (unsigned short)(u >> 16);
}
__device__ __forceinline__ u32 fenc(float f){
    u32 u = __float_as_uint(f);
    return (u & 0x80000000u) ? ~u : (u | 0x80000000u);
}
__device__ __forceinline__ float fdec(u32 u){
    return (u & 0x80000000u) ? __uint_as_float(u & 0x7fffffffu) : __uint_as_float(~u);
}
__device__ __forceinline__ float sigm(float x){ return 1.f / (1.f + __expf(-x)); }

__device__ __forceinline__ void edge_sd(const int* ei, int e, int E, int& s, int& d){
    if (e < E){ s = ei[e]; d = ei[E + e]; } else { s = e - E; d = s; }
}

// ---------------------------------------------------------------------------
// Pack fp32 activations [M x K] -> WMMA-A bf16 packed layout:
//   index = ((mt*Kt + kt)*32 + lane)*8 + v ; lane: m=lane&15, hf=lane>>4
//   k = kt*32 + (v&4?16:0) + 8*hf + 2*(v&3) (+ pair)
// ---------------------------------------------------------------------------
__global__ __launch_bounds__(256) void k_pack_a(const float* __restrict__ src,
        u32* __restrict__ dst, int K, int Kp, size_t total)
{
    size_t i = (size_t)blockIdx.x * 256 + threadIdx.x;
    if (i >= total) return;
    int v    = (int)(i & 7);
    int lane = (int)((i >> 3) & 31);
    size_t t = i >> 8;
    int Ktt  = Kp >> 5;
    int kt   = (int)(t % Ktt);
    int mt   = (int)(t / Ktt);
    int m = lane & 15, hf = lane >> 4;
    int k0 = (kt << 5) + ((v & 4) ? 16 : 0) + (hf << 3) + ((v & 3) << 1);
    size_t row = (size_t)((mt << 4) + m);
    float f0 = (k0     < K) ? src[row * K + k0    ] : 0.f;
    float f1 = (k0 + 1 < K) ? src[row * K + k0 + 1] : 0.f;
    dst[i] = (u32)bf16r(f0) | ((u32)bf16r(f1) << 16);
}

// ---------------------------------------------------------------------------
// Pack fp32 weights W [Nout x K] (PyTorch layout, used as B = W^T, K x Nout)
// into WMMA-B bf16 packed layout:
//   index = ((kt*Ntt + nt)*32 + lane)*8 + v ; n = nt*16 + (lane&15)
//   k = kt*32 + (v&4?16:0) + 8*(lane>>4) + 2*(v&3) (+ pair)
// ---------------------------------------------------------------------------
__global__ __launch_bounds__(256) void k_pack_b(const float* __restrict__ W,
        u32* __restrict__ dst, int Ntt, int K, size_t total)
{
    size_t i = (size_t)blockIdx.x * 256 + threadIdx.x;
    if (i >= total) return;
    int v    = (int)(i & 7);
    int lane = (int)((i >> 3) & 31);
    size_t t = i >> 8;
    int nt = (int)(t % Ntt);
    int kt = (int)(t / Ntt);
    int n  = (nt << 4) + (lane & 15);
    int kh = lane >> 4;
    int k0 = (kt << 5) + ((v & 4) ? 16 : 0) + (kh << 3) + ((v & 3) << 1);
    float f0 = (k0     < K) ? W[(size_t)n * K + k0    ] : 0.f;
    float f1 = (k0 + 1 < K) ? W[(size_t)n * K + k0 + 1] : 0.f;
    dst[i] = (u32)bf16r(f0) | ((u32)bf16r(f1) << 16);
}

// ---------------------------------------------------------------------------
// bf16 WMMA GEMM: C[M x Nc] = A(bf16 packed, M x Kp) * B(bf16 packed, Kp x Nc)
// One wave -> 16x64 strip (4 accumulators, A fragment reused across 4 WMMAs).
// Block of 4 waves -> 16x256 strip; Nc must be a multiple of 256.
// ---------------------------------------------------------------------------
__global__ __launch_bounds__(128) void k_gemm(const u32* __restrict__ Ap,
        const u32* __restrict__ Bp, float* __restrict__ C,
        const float* __restrict__ bias, int Kt, int Nc, int accF)
{
    const int lane = threadIdx.x & 31;
    const int wave = threadIdx.x >> 5;
    const int mt = blockIdx.x;
    const int nt0 = (blockIdx.y * 4 + wave) * 4;   // first of 4 16-col tiles
    const int ntiles = Nc >> 4;
    if (nt0 >= ntiles) return;
    const u32* ap = Ap + ((size_t)mt * Kt) * 256 + lane * 8;
    const u32* bp = Bp + (size_t)nt0 * 256 + lane * 8;
    const size_t bstep = (size_t)ntiles * 256;
    v8f acc0 = {0.f,0.f,0.f,0.f,0.f,0.f,0.f,0.f};
    v8f acc1 = acc0, acc2 = acc0, acc3 = acc0;
    for (int kt = 0; kt < Kt; ++kt){
        union { uint4 q[2]; v16bf v; } ua, ub0, ub1, ub2, ub3;
        ua.q[0]  = *(const uint4*)(ap);
        ua.q[1]  = *(const uint4*)(ap + 4);
        ub0.q[0] = *(const uint4*)(bp);
        ub0.q[1] = *(const uint4*)(bp + 4);
        ub1.q[0] = *(const uint4*)(bp + 256);
        ub1.q[1] = *(const uint4*)(bp + 260);
        ub2.q[0] = *(const uint4*)(bp + 512);
        ub2.q[1] = *(const uint4*)(bp + 516);
        ub3.q[0] = *(const uint4*)(bp + 768);
        ub3.q[1] = *(const uint4*)(bp + 772);
        __builtin_prefetch(ap + 256, 0, 3);
        __builtin_prefetch(bp + bstep, 0, 3);
        __builtin_prefetch(bp + bstep + 512, 0, 3);
        ap += 256; bp += bstep;
        acc0 = __builtin_amdgcn_wmma_f32_16x16x32_bf16(
                   false, ua.v, false, ub0.v, (short)0, acc0, false, false);
        acc1 = __builtin_amdgcn_wmma_f32_16x16x32_bf16(
                   false, ua.v, false, ub1.v, (short)0, acc1, false, false);
        acc2 = __builtin_amdgcn_wmma_f32_16x16x32_bf16(
                   false, ua.v, false, ub2.v, (short)0, acc2, false, false);
        acc3 = __builtin_amdgcn_wmma_f32_16x16x32_bf16(
                   false, ua.v, false, ub3.v, (short)0, acc3, false, false);
    }
    // C/D layout: VGPR j: lanes0-15 -> M=j, lanes16-31 -> M=j+8 ; N=lane&15
    const int m0 = (mt << 4) + ((lane >> 4) << 3);
    #pragma unroll
    for (int tile = 0; tile < 4; ++tile){
        const v8f* accp = (tile == 0) ? &acc0 : (tile == 1) ? &acc1
                        : (tile == 2) ? &acc2 : &acc3;
        const int n = ((nt0 + tile) << 4) + (lane & 15);
        const float bv = bias ? bias[n] : 0.f;
        #pragma unroll
        for (int j = 0; j < 8; ++j){
            size_t idx = (size_t)(m0 + j) * Nc + n;
            float v = (*accp)[j] + bv;
            if (accF) v += C[idx];
            C[idx] = v;
        }
    }
}

// ---------------------------------------------------------------------------
__global__ __launch_bounds__(256) void k_zero(float* __restrict__ p, size_t n){
    size_t i = (size_t)blockIdx.x * 256 + threadIdx.x;
    if (i < n) p[i] = 0.f;
}

__global__ __launch_bounds__(256) void k_init_md(u32* __restrict__ mmax,
        float* __restrict__ den, int n){
    int i = blockIdx.x * 256 + threadIdx.x;
    if (i < n){ mmax[i] = 0x007FFFFFu; den[i] = 0.f; }  // enc(-inf)
}

// LSTM elementwise: gates [N x 1024] (i,f,g,o), updates h,c (fp32)
__global__ __launch_bounds__(256) void k_lstm(const float* __restrict__ gates,
        const float* __restrict__ bih, const float* __restrict__ bhh,
        float* __restrict__ h, float* __restrict__ c, size_t total)
{
    size_t t = (size_t)blockIdx.x * 256 + threadIdx.x;
    if (t >= total) return;
    size_t n = t >> 8;
    int j = (int)(t & 255);
    const float* g = gates + (n << 10);
    float gi = g[j]       + bih[j]       + bhh[j];
    float gf = g[256 + j] + bih[256 + j] + bhh[256 + j];
    float gg = g[512 + j] + bih[512 + j] + bhh[512 + j];
    float go = g[768 + j] + bih[768 + j] + bhh[768 + j];
    float cv = sigm(gf) * c[t] + sigm(gi) * tanhf(gg);
    float hv = sigm(go) * tanhf(cv);
    c[t] = cv; h[t] = hv;
}

// GATv2 edge logits: per (edge, head) dot(att_h, leaky_relu(xl[src]+xr[dst]))
__global__ __launch_bounds__(256) void k_logits(const float* __restrict__ xl,
        const float* __restrict__ xr, const int* __restrict__ ei,
        const float* __restrict__ att, float* __restrict__ logit, int E, int Etot)
{
    int t = blockIdx.x * 256 + threadIdx.x;
    if (t >= Etot * NHEADS) return;
    int e = t >> 2, hd = t & 3;
    int sN, dN; edge_sd(ei, e, E, sN, dN);
    const float* pl = xl + ((size_t)sN << 8) + (hd << 6);
    const float* pr = xr + ((size_t)dN << 8) + (hd << 6);
    const float* pa = att + (hd << 6);
    float a = 0.f;
    for (int cc = 0; cc < 64; ++cc){
        float v = pl[cc] + pr[cc];
        v = (v > 0.f) ? v : 0.2f * v;
        a += pa[cc] * v;
    }
    logit[t] = a;
}

__global__ __launch_bounds__(256) void k_segmax(const float* __restrict__ logit,
        const int* __restrict__ ei, u32* __restrict__ mmax, int E, int Etot)
{
    int t = blockIdx.x * 256 + threadIdx.x;
    if (t >= Etot * NHEADS) return;
    int e = t >> 2, hd = t & 3;
    int sN, dN; edge_sd(ei, e, E, sN, dN);
    atomicMax(&mmax[dN * 4 + hd], fenc(logit[t]));
}

__global__ __launch_bounds__(256) void k_expden(const float* __restrict__ logit,
        const int* __restrict__ ei, const u32* __restrict__ mmax,
        float* __restrict__ den, float* __restrict__ aexp, int E, int Etot)
{
    int t = blockIdx.x * 256 + threadIdx.x;
    if (t >= Etot * NHEADS) return;
    int e = t >> 2, hd = t & 3;
    int sN, dN; edge_sd(ei, e, E, sN, dN);
    float a = __expf(logit[t] - fdec(mmax[dN * 4 + hd]));
    aexp[t] = a;
    atomicAdd(&den[dN * 4 + hd], a);
}

__global__ __launch_bounds__(256) void k_scatter(const float* __restrict__ xl,
        const int* __restrict__ ei, const float* __restrict__ aexp,
        const float* __restrict__ den, float* __restrict__ outr,
        int E, long long total)
{
    long long t = (long long)blockIdx.x * 256 + threadIdx.x;
    if (t >= total) return;
    int e = (int)(t >> 8);
    int j = (int)(t & 255);
    int hd = j >> 6;
    int sN, dN; edge_sd(ei, e, E, sN, dN);
    float alpha = aexp[(size_t)e * 4 + hd] / den[(size_t)dN * 4 + hd];
    atomicAdd(&outr[((size_t)dN << 8) + j], alpha * xl[((size_t)sN << 8) + j]);
}

// LayerNorm over a row, modes: 0 = (+addb) LN -> ELU -> pack bf16
//                              1 = LN -> ReLU -> pack bf16
//                              2 = LN -> fp32 out
__global__ __launch_bounds__(256) void k_ln(const float* __restrict__ src,
        const float* __restrict__ addb, const float* __restrict__ g,
        const float* __restrict__ b, int W, int mode,
        u32* __restrict__ dstp, float* __restrict__ dstf)
{
    const int row = blockIdx.x;
    const float* s = src + (size_t)row * W;
    __shared__ float red[256];
    float vals[3];
    const int per = (W + 255) >> 8;
    float acc = 0.f;
    for (int r = 0; r < per; ++r){
        int j = threadIdx.x + (r << 8);
        float v = 0.f;
        if (j < W){ v = s[j]; if (addb) v += addb[j]; }
        vals[r] = v; acc += v;
    }
    red[threadIdx.x] = acc; __syncthreads();
    for (int o = 128; o > 0; o >>= 1){
        if (threadIdx.x < o) red[threadIdx.x] += red[threadIdx.x + o];
        __syncthreads();
    }
    const float mean = red[0] / W;
    __syncthreads();
    acc = 0.f;
    for (int r = 0; r < per; ++r){
        int j = threadIdx.x + (r << 8);
        if (j < W){ float d = vals[r] - mean; acc += d * d; }
    }
    red[threadIdx.x] = acc; __syncthreads();
    for (int o = 128; o > 0; o >>= 1){
        if (threadIdx.x < o) red[threadIdx.x] += red[threadIdx.x + o];
        __syncthreads();
    }
    const float rstd = rsqrtf(red[0] / W + 1e-5f);
    for (int r = 0; r < per; ++r){
        int j = threadIdx.x + (r << 8);
        if (j >= W) continue;
        float y = (vals[r] - mean) * rstd * g[j] + b[j];
        if (mode == 0)      y = (y > 0.f) ? y : (__expf(y) - 1.f);
        else if (mode == 1) y = fmaxf(y, 0.f);
        if (mode <= 1){
            // pack into WMMA-A bf16 layout (Kpad == W)
            int mt = row >> 4, m = row & 15;
            int kt = j >> 5, kk = j & 31;
            int hf = (kk >> 3) & 1;
            int lane = m + (hf << 4);
            int v = ((kk & 16) ? 4 : 0) + ((kk >> 1) & 3);
            size_t ui = ((((size_t)mt * (W >> 5) + kt) << 5) + lane) * 8 + v;
            ((unsigned short*)dstp)[(ui << 1) + (kk & 1)] = bf16r(y);
        } else {
            dstf[(size_t)row * W + j] = y;
        }
    }
}

__global__ __launch_bounds__(256) void k_pool(const float* __restrict__ h,
        const int* __restrict__ batch, float* __restrict__ pooled,
        float* __restrict__ cnt, size_t total)
{
    size_t t = (size_t)blockIdx.x * 256 + threadIdx.x;
    if (t >= total) return;
    size_t n = t >> 8;
    int j = (int)(t & 255);
    int gi = batch[n];
    atomicAdd(&pooled[(size_t)gi * 256 + j], h[t]);
    if (j == 0) atomicAdd(&cnt[gi], 1.f);
}

__global__ __launch_bounds__(256) void k_pooldiv(float* __restrict__ pooled,
        const float* __restrict__ cnt, size_t total)
{
    size_t t = (size_t)blockIdx.x * 256 + threadIdx.x;
    if (t >= total) return;
    pooled[t] /= fmaxf(cnt[t >> 8], 1.f);
}

// ---------------------------------------------------------------------------
extern "C" void kernel_launch(void* const* d_in, const int* in_sizes, int n_in,
                              void* d_out, int out_size, void* d_ws, size_t ws_size,
                              hipStream_t stream)
{
    (void)n_in; (void)ws_size;
    const float* x  = (const float*)d_in[0];
    const int* ei   = (const int*)d_in[1];
    const int* bat  = (const int*)d_in[2];
    const int N   = in_sizes[2];
    const int E   = in_sizes[1] / 2;
    const int FIN = in_sizes[0] / N;
    const int H = 256, NHID = 512, NOUT = 768;
    const int Bg = out_size / NOUT;
    const int FINp = (FIN + 31) & ~31;
    const int Etot = E + N;

    char* wsb = (char*)d_ws;
    size_t off = 0;
    auto alloc = [&](size_t bytes) -> void* {
        void* p = wsb + off;
        off = (off + bytes + 255) & ~(size_t)255;
        return p;
    };

    float* gates = (float*)alloc((size_t)N * 4 * H * 4);
    float* xl    = gates;                       // alias: xl/xr live before gates
    float* xr    = gates + (size_t)N * H;
    float* outr  = (float*)alloc((size_t)N * H * 4);
    float* hbuf  = (float*)alloc((size_t)N * H * 4);
    float* cbuf  = (float*)alloc((size_t)N * H * 4);
    u32* xb  = (u32*)alloc((size_t)N * FINp * 2);
    u32* yb  = (u32*)alloc((size_t)N * H * 2);
    u32* hb  = (u32*)alloc((size_t)N * H * 2);
    float* logit = (float*)alloc((size_t)Etot * 4 * 4);
    float* aexp  = (float*)alloc((size_t)Etot * 4 * 4);
    u32* mmax    = (u32*)alloc((size_t)N * 4 * 4);
    float* den   = (float*)alloc((size_t)N * 4 * 4);
    float* pooled = (float*)alloc((size_t)Bg * H * 4);
    float* cnt    = (float*)alloc((size_t)Bg * 4);
    u32* pooledb  = (u32*)alloc((size_t)Bg * H * 2);
    float* z1     = (float*)alloc((size_t)Bg * NHID * 4);
    u32* z1b      = (u32*)alloc((size_t)Bg * NHID * 2);
    float* z2     = (float*)alloc((size_t)Bg * NOUT * 4);

    auto wballoc = [&](int Nout, int Kp) -> u32* {
        return (u32*)alloc((size_t)Kp * Nout * 2);
    };
    u32* lin0B = wballoc(H, FINp);
    u32* rnn0B = wballoc(4 * H, H);
    u32 *wlB[3], *wrB[3], *wihB[3], *whhB[3];
    for (int L = 0; L < 3; ++L){
        int fp = (L == 0) ? FINp : H;
        wlB[L]  = wballoc(H, fp);
        wrB[L]  = wballoc(H, fp);
        wihB[L] = wballoc(4 * H, H);
        whhB[L] = wballoc(4 * H, H);
    }
    u32* mh1B = wballoc(NHID, H);
    u32* mh2B = wballoc(NOUT, NHID);

    auto packA = [&](const float* src, u32* dst, int M, int K, int Kp){
        size_t tot = (size_t)(M >> 4) * (Kp >> 5) * 256;
        k_pack_a<<<(unsigned)((tot + 255) >> 8), 256, 0, stream>>>(src, dst, K, Kp, tot);
    };
    auto packB = [&](const float* W, u32* dst, int Nout, int K, int Kp){
        size_t tot = (size_t)(Kp >> 5) * (Nout >> 4) * 256;
        k_pack_b<<<(unsigned)((tot + 255) >> 8), 256, 0, stream>>>(W, dst, Nout >> 4, K, tot);
    };
    auto gemm = [&](const u32* Ap, const u32* Bp, float* C, const float* bias,
                    int M, int Kp, int Nc, int accF){
        dim3 grid((unsigned)(M >> 4), (unsigned)(Nc >> 8));
        k_gemm<<<grid, 128, 0, stream>>>(Ap, Bp, C, bias, Kp >> 5, Nc, accF);
    };
    auto zero = [&](float* p, size_t n){
        k_zero<<<(unsigned)((n + 255) >> 8), 256, 0, stream>>>(p, n);
    };

    // ---- pack all weights to WMMA-B bf16 ----
    packB((const float*)d_in[4], lin0B, H, FIN, FINp);
    packB((const float*)d_in[6], rnn0B, 4 * H, H, H);
    for (int L = 0; L < 3; ++L){
        int bi = 10 + L * 12;
        int fin = (L == 0) ? FIN : H;
        int fp  = (L == 0) ? FINp : H;
        packB((const float*)d_in[bi + 0], wlB[L],  H, fin, fp);
        packB((const float*)d_in[bi + 2], wrB[L],  H, fin, fp);
        packB((const float*)d_in[bi + 8], wihB[L], 4 * H, H, H);
        packB((const float*)d_in[bi + 9], whhB[L], 4 * H, H, H);
    }
    packB((const float*)d_in[46], mh1B, NHID, H, H);
    packB((const float*)d_in[50], mh2B, NOUT, NHID, NHID);

    // ---- stage 0: lin0 + rnn0 (h=0, c=0 -> skip whh matmul) ----
    packA(x, xb, N, FIN, FINp);
    gemm(xb, lin0B, outr, (const float*)d_in[5], N, FINp, H, 0);
    packA(outr, yb, N, H, H);
    gemm(yb, rnn0B, gates, nullptr, N, H, 4 * H, 0);
    zero(cbuf, (size_t)N * H);
    {
        size_t tot = (size_t)N * H;
        k_lstm<<<(unsigned)((tot + 255) >> 8), 256, 0, stream>>>(
            gates, (const float*)d_in[8], (const float*)d_in[9], hbuf, cbuf, tot);
    }
    packA(hbuf, hb, N, H, H);

    // ---- 3 x (GATv2 -> LN -> ELU -> LSTM) ----
    for (int L = 0; L < 3; ++L){
        int bi = 10 + L * 12;
        const u32* xin = (L == 0) ? xb : hb;
        int Kp = (L == 0) ? FINp : H;
        gemm(xin, wlB[L], xl, (const float*)d_in[bi + 1], N, Kp, H, 0);
        gemm(xin, wrB[L], xr, (const float*)d_in[bi + 3], N, Kp, H, 0);
        zero(outr, (size_t)N * H);
        k_init_md<<<(unsigned)((N * 4 + 255) >> 8), 256, 0, stream>>>(mmax, den, N * 4);
        int te = Etot * 4;
        unsigned gbe = (unsigned)((te + 255) >> 8);
        k_logits<<<gbe, 256, 0, stream>>>(xl, xr, ei, (const float*)d_in[bi + 4], logit, E, Etot);
        k_segmax<<<gbe, 256, 0, stream>>>(logit, ei, mmax, E, Etot);
        k_expden<<<gbe, 256, 0, stream>>>(logit, ei, mmax, den, aexp, E, Etot);
        long long ts = (long long)Etot * H;
        k_scatter<<<(unsigned)((ts + 255) >> 8), 256, 0, stream>>>(xl, ei, aexp, den, outr, E, ts);
        // +bias, LN, ELU, pack bf16 for the LSTM input GEMM
        k_ln<<<(unsigned)N, 256, 0, stream>>>(outr, (const float*)d_in[bi + 5],
            (const float*)d_in[bi + 6], (const float*)d_in[bi + 7], H, 0, yb, nullptr);
        gemm(yb, wihB[L], gates, nullptr, N, H, 4 * H, 0);
        gemm(hb, whhB[L], gates, nullptr, N, H, 4 * H, 1);
        size_t tot = (size_t)N * H;
        k_lstm<<<(unsigned)((tot + 255) >> 8), 256, 0, stream>>>(
            gates, (const float*)d_in[bi + 10], (const float*)d_in[bi + 11], hbuf, cbuf, tot);
        packA(hbuf, hb, N, H, H);
    }

    // ---- mean pool over graphs ----
    zero(pooled, (size_t)Bg * H);
    zero(cnt, (size_t)Bg);
    {
        size_t tot = (size_t)N * H;
        k_pool<<<(unsigned)((tot + 255) >> 8), 256, 0, stream>>>(hbuf, bat, pooled, cnt, tot);
    }
    k_pooldiv<<<(unsigned)(((size_t)Bg * H + 255) >> 8), 256, 0, stream>>>(
        pooled, cnt, (size_t)Bg * H);
    packA(pooled, pooledb, Bg, H, H);

    // ---- output MLP ----
    gemm(pooledb, mh1B, z1, (const float*)d_in[47], Bg, H, NHID, 0);
    k_ln<<<(unsigned)Bg, 256, 0, stream>>>(z1, nullptr, (const float*)d_in[48],
        (const float*)d_in[49], NHID, 1, z1b, nullptr);
    gemm(z1b, mh2B, z2, (const float*)d_in[51], Bg, NHID, NOUT, 0);
    k_ln<<<(unsigned)Bg, 256, 0, stream>>>(z2, nullptr, (const float*)d_in[52],
        (const float*)d_in[53], NOUT, 2, nullptr, (float*)d_out);
}
